// CrissCrossAttention_4690104287722
// MI455X (gfx1250) — matrix operand order repaired and checked by reading, not compiled, not run on hardware
//
#include <hip/hip_runtime.h>
#include <math.h>

// ---------------------------------------------------------------------------
// CDNA5 (gfx1250) criss-cross attention, wave32 + v_wmma_f32_16x16x32_bf16.
// B=16, C=64, H=W=128, C8=8.
// ---------------------------------------------------------------------------

typedef __bf16 bf16_t;
typedef __attribute__((ext_vector_type(16))) __bf16 v16bf;
typedef __attribute__((ext_vector_type(8)))  float  v8f;

#define WMMA_BF16(A, B, C) \
  __builtin_amdgcn_wmma_f32_16x16x32_bf16(false, (A), false, (B), (short)0, (C), false, false)

union AFrag { v16bf v; unsigned u[8]; };
union BFrag { v16bf v; unsigned u[8]; };
union Pack2 { bf16_t h[2]; unsigned u; };

// A (16x32 bf16): lane L holds row M = mBase + (L%16).
// elements: K = (L>=16 ? 16 : 0) + e, packed pairs => 8 aligned b32 LDS loads.
__device__ __forceinline__ void loadA16(AFrag& f, const bf16_t* base, int stride,
                                        int mBase, int kBase, int lane) {
  const bf16_t* row = base + (mBase + (lane & 15)) * stride;
  int kb = kBase + ((lane >> 4) & 1) * 16;
#pragma unroll
  for (int j = 0; j < 8; ++j) {
    int k = kb + 2 * j;
    f.u[j] = *(const unsigned*)(row + k);
  }
}

// B (32x16 bf16) staged as [N][K]: lane L holds col N = nBase + (L%16),
// K = (L>=16 ? 16 : 0) + e  => contiguous pairs, 8 aligned b32 LDS loads.
__device__ __forceinline__ void loadB16(BFrag& f, const bf16_t* base, int stride,
                                        int nBase, int kBase, int lane) {
  const bf16_t* row = base + (nBase + (lane & 15)) * stride;
  int kb = kBase + ((lane >> 4) & 1) * 16;
#pragma unroll
  for (int j = 0; j < 8; ++j) f.u[j] = *(const unsigned*)(row + kb + 2 * j);
}

// ---------------------------------------------------------------------------
// Kernel 0: fused QKV projection.  Per batch: [Wq;Wk;Wv] (80x64) x X (64x16384).
// grid = B*64 blocks, each block does a 256-pixel N-tile; 256 threads = 8 waves.
// Loop order (mt outer, compile-time) keeps the q/k/v store target uniform:
// mt>=1 stores are straight v stores; only mt==0 needs a q/k pointer select.
// ---------------------------------------------------------------------------
__global__ __launch_bounds__(256) void proj_kernel(
    const float* __restrict__ x,
    const float* __restrict__ Wq, const float* __restrict__ bq,
    const float* __restrict__ Wk, const float* __restrict__ bk,
    const float* __restrict__ Wv, const float* __restrict__ bv,
    bf16_t* __restrict__ qO, bf16_t* __restrict__ kO, bf16_t* __restrict__ vO) {
  __shared__ bf16_t Wl[80][66];    // [M][K], odd word stride -> conflict-free
  __shared__ bf16_t XT[256][66];   // [N][K]
  __shared__ float  bias[80];

  const int b = blockIdx.x >> 6;
  const int pix0 = (blockIdx.x & 63) << 8;
  const int t = threadIdx.x;

  for (int i = t; i < 80 * 64; i += 256) {
    int m = i >> 6, c = i & 63;
    float w = (m < 8) ? Wq[m * 64 + c] : (m < 16) ? Wk[(m - 8) * 64 + c] : Wv[(m - 16) * 64 + c];
    Wl[m][c] = (bf16_t)w;
  }
  if (t < 80) bias[t] = (t < 8) ? bq[t] : (t < 16) ? bk[t - 8] : bv[t - 16];
#pragma unroll 4
  for (int c = 0; c < 64; ++c)
    XT[t][c] = (bf16_t)x[(size_t)(b * 64 + c) * 16384 + pix0 + t];
  __syncthreads();

  const int lane = t & 31, wv = t >> 5;
#pragma unroll
  for (int mt = 0; mt < 5; ++mt) {
    for (int nt = wv; nt < 16; nt += 8) {   // 2 n-tiles per wave per mt
      v8f acc = {};
#pragma unroll
      for (int ks = 0; ks < 2; ++ks) {
        AFrag a; loadA16(a, &Wl[0][0], 66, mt * 16, ks * 32, lane);
        BFrag bb; loadB16(bb, &XT[0][0], 66, nt * 16, ks * 32, lane);
        acc = WMMA_BF16(a.v, bb.v, acc);
      }
      const size_t pix = (size_t)pix0 + nt * 16 + (lane & 15);
      const int mb = mt * 16 + ((lane >> 4) << 3);
#pragma unroll
      for (int r = 0; r < 8; ++r) {
        int m = mb + r;
        bf16_t hv = (bf16_t)(acc[r] + bias[m]);
        if (mt == 0) {
          // m in [0,16): q for m<8, k for m-8 otherwise (uniform per half-wave)
          bf16_t* dst = (m < 8) ? (qO + ((size_t)(b * 8 + m) << 14))
                                : (kO + ((size_t)(b * 8 + m - 8) << 14));
          dst[pix] = hv;
        } else {
          vO[((size_t)(b * 64 + m - 16) << 14) + pix] = hv;
        }
      }
    }
  }
}

// ---------------------------------------------------------------------------
// Kernel 1: bf16 128x128 plane transpose (row-major -> col-major copies).
// grid = planes*16 (32x32 tiles), 256 threads (32x8).
// ---------------------------------------------------------------------------
__global__ __launch_bounds__(256) void transpose_kernel(
    const bf16_t* __restrict__ src, bf16_t* __restrict__ dst) {
  __shared__ bf16_t tl[32][34];
  const int plane = blockIdx.x >> 4;
  const int tid = blockIdx.x & 15;
  const int r0 = (tid >> 2) << 5, c0 = (tid & 3) << 5;
  const int tx = threadIdx.x & 31, ty = threadIdx.x >> 5;
  const size_t base = (size_t)plane << 14;
#pragma unroll
  for (int i = ty; i < 32; i += 8) tl[i][tx] = src[base + (size_t)(r0 + i) * 128 + c0 + tx];
  __syncthreads();
#pragma unroll
  for (int i = ty; i < 32; i += 8) dst[base + (size_t)(c0 + i) * 128 + r0 + tx] = tl[tx][i];
}

// ---------------------------------------------------------------------------
// Kernel 2: one attention "line" (a row h in pass W, a column w in pass H).
//   E(128x128) = q_line^T x k_line (K=8 zero-padded to 32) via WMMA
//   per-pixel max / exp / sum (split-softmax stats), optional diagonal mask
//   probs re-encoded bf16 into P, then outUn(64x128) = V x P^T via WMMA (K=128)
// grid = B*128 blocks, 256 threads = 8 waves.
// ---------------------------------------------------------------------------
__global__ __launch_bounds__(256) void attn_line_kernel(
    const bf16_t* __restrict__ q, const bf16_t* __restrict__ k,
    const bf16_t* __restrict__ v, float* __restrict__ outUn,
    float* __restrict__ mOut, float* __restrict__ sOut, int maskDiag) {
  __shared__ bf16_t qT[128][34];   // [pixel][c], c 8..33 zeroed (K pad)
  __shared__ bf16_t kT[128][34];   // [attended][c]
  __shared__ bf16_t vT[64][130];   // [c][attended]
  __shared__ float  E[128][129];   // raw energies
  __shared__ bf16_t P[128][130];   // softmax numerator, bf16  [pixel][attended]

  const int b = blockIdx.x >> 7, line = blockIdx.x & 127;
  const int t = threadIdx.x;

  if (t < 128) {
#pragma unroll
    for (int c = 0; c < 8; ++c) {
      qT[t][c] = q[(size_t)(b * 8 + c) * 16384 + line * 128 + t];
      kT[t][c] = k[(size_t)(b * 8 + c) * 16384 + line * 128 + t];
    }
#pragma unroll
    for (int c = 8; c < 34; ++c) { qT[t][c] = (bf16_t)0.f; kT[t][c] = (bf16_t)0.f; }
  }
  for (int i = t; i < 64 * 128; i += 256) {
    int c = i >> 7, j = i & 127;
    vT[c][j] = v[(size_t)(b * 64 + c) * 16384 + line * 128 + j];
  }
  __syncthreads();

  const int lane = t & 31, wv = t >> 5;

  // ---- GEMM 1: energies (wave wv owns M-strip wv, all 8 N-tiles) ----
  {
    AFrag aq; loadA16(aq, &qT[0][0], 34, wv * 16, 0, lane);   // invariant per wave
#pragma unroll
    for (int nt = 0; nt < 8; ++nt) {
      v8f acc = {};
      BFrag bb; loadB16(bb, &kT[0][0], 34, nt * 16, 0, lane);
      acc = WMMA_BF16(aq.v, bb.v, acc);
      int col = nt * 16 + (lane & 15);
      int rowb = wv * 16 + ((lane >> 4) << 3);
#pragma unroll
      for (int r = 0; r < 8; ++r) E[rowb + r][col] = acc[r];
    }
  }
  __syncthreads();

  // ---- split softmax stats (thread t owns pixel row t) ----
  if (t < 128) {
    if (maskDiag) E[t][t] = -INFINITY;   // energy_H diagonal mask
    float m = -INFINITY;
    for (int j = 0; j < 128; ++j) m = fmaxf(m, E[t][j]);
    float s = 0.f;
    for (int j = 0; j < 128; ++j) {
      float p = __expf(E[t][j] - m);
      P[t][j] = (bf16_t)p;
      s += p;
    }
    mOut[((size_t)b << 14) + line * 128 + t] = m;
    sOut[((size_t)b << 14) + line * 128 + t] = s;
  }
  __syncthreads();

  // ---- GEMM 2: outUn = V x P^T  (4 M-tiles x 8 N-tiles; 4 tiles per wave) ----
  const int mt = wv & 3, nt0 = (wv >> 2) << 2;
#pragma unroll
  for (int nt = nt0; nt < nt0 + 4; ++nt) {
    v8f acc = {};
#pragma unroll
    for (int ks = 0; ks < 4; ++ks) {
      AFrag a; loadA16(a, &vT[0][0], 130, mt * 16, ks * 32, lane);
      BFrag bb; loadB16(bb, &P[0][0], 130, nt * 16, ks * 32, lane);
      acc = WMMA_BF16(a.v, bb.v, acc);
    }
    int col = nt * 16 + (lane & 15);
    int rowb = mt * 16 + ((lane >> 4) << 3);
#pragma unroll
    for (int r = 0; r < 8; ++r)
      outUn[(size_t)(b * 64 + rowb + r) * 16384 + line * 128 + col] = acc[r];
  }
}

// ---------------------------------------------------------------------------
// Kernel 3: merge softmax stats of the two branches -> per-pixel scale factors
// (gamma folded in).  scale* stored row-major (B,H,W).
// ---------------------------------------------------------------------------
__global__ __launch_bounds__(256) void scales_kernel(
    const float* __restrict__ mW, const float* __restrict__ sW,
    const float* __restrict__ mH, const float* __restrict__ sH,
    const float* __restrict__ gamma,
    float* __restrict__ scW, float* __restrict__ scH) {
  int idx = blockIdx.x * 256 + threadIdx.x;          // B*H*W = 262144
  int b = idx >> 14, r = idx & 16383;
  int h = r >> 7, w = r & 127;
  size_t cmIdx = ((size_t)b << 14) + w * 128 + h;    // col-major stats
  float mw = mW[idx], sw = sW[idx];
  float mh = mH[cmIdx], sh = sH[cmIdx];
  float m = fmaxf(mw, mh);
  float ew = __expf(mw - m), eh = __expf(mh - m);
  float inv = gamma[0] / (sw * ew + sh * eh);
  scW[idx] = ew * inv;
  scH[idx] = eh * inv;
}

// ---------------------------------------------------------------------------
// Kernel 4: out = oH(col-major, transposed via LDS)*scH + oW*scW + x
// grid = B*64 planes * 16 tiles, 256 threads (32x8).
// ---------------------------------------------------------------------------
__global__ __launch_bounds__(256) void combine_kernel(
    const float* __restrict__ oW, const float* __restrict__ oH_cm,
    const float* __restrict__ scW, const float* __restrict__ scH,
    const float* __restrict__ x, float* __restrict__ out) {
  __shared__ float tl[32][33];
  const int plane = blockIdx.x >> 4;                 // b*64 + c
  const int tid = blockIdx.x & 15;
  const int h0 = (tid >> 2) << 5, w0 = (tid & 3) << 5;
  const int b = plane >> 6;
  const int tx = threadIdx.x & 31, ty = threadIdx.x >> 5;
  const size_t pbase = (size_t)plane << 14;
  const size_t sbase = (size_t)b << 14;
  // tl[i][j] = oH(h = h0+j, w = w0+i)
#pragma unroll
  for (int i = ty; i < 32; i += 8) tl[i][tx] = oH_cm[pbase + (size_t)(w0 + i) * 128 + h0 + tx];
  __syncthreads();
#pragma unroll
  for (int i = ty; i < 32; i += 8) {
    int h = h0 + i, w = w0 + tx;
    size_t gi = pbase + (size_t)h * 128 + w;
    size_t si = sbase + (size_t)h * 128 + w;
    out[gi] = tl[tx][i] * scH[si] + oW[gi] * scW[si] + x[gi];
  }
}

// ---------------------------------------------------------------------------
extern "C" void kernel_launch(void* const* d_in, const int* in_sizes, int n_in,
                              void* d_out, int out_size, void* d_ws, size_t ws_size,
                              hipStream_t stream) {
  (void)in_sizes; (void)n_in; (void)out_size; (void)ws_size;
  const float* x     = (const float*)d_in[0];
  const float* Wq    = (const float*)d_in[1];
  const float* bq    = (const float*)d_in[2];
  const float* Wk    = (const float*)d_in[3];
  const float* bk    = (const float*)d_in[4];
  const float* Wv    = (const float*)d_in[5];
  const float* bv    = (const float*)d_in[6];
  const float* gamma = (const float*)d_in[7];
  float* out = (float*)d_out;

  const size_t NP = (size_t)16 * 16384;  // pixels
  char* ws = (char*)d_ws;
  size_t off = 0;
  auto alloc = [&](size_t bytes) { void* p = ws + off; off = (off + bytes + 255) & ~(size_t)255; return p; };

  bf16_t* q_rm = (bf16_t*)alloc(NP * 8 * 2);
  bf16_t* k_rm = (bf16_t*)alloc(NP * 8 * 2);
  bf16_t* v_rm = (bf16_t*)alloc(NP * 64 * 2);
  bf16_t* q_cm = (bf16_t*)alloc(NP * 8 * 2);
  bf16_t* k_cm = (bf16_t*)alloc(NP * 8 * 2);
  bf16_t* v_cm = (bf16_t*)alloc(NP * 64 * 2);
  float*  oW   = (float*)alloc(NP * 64 * 4);
  float*  oH   = (float*)alloc(NP * 64 * 4);
  float*  mW   = (float*)alloc(NP * 4);
  float*  sW   = (float*)alloc(NP * 4);
  float*  mH   = (float*)alloc(NP * 4);
  float*  sH   = (float*)alloc(NP * 4);
  float*  scW  = (float*)alloc(NP * 4);
  float*  scH  = (float*)alloc(NP * 4);

  dim3 blk(256);
  // 1) QKV projection (WMMA GEMM)
  proj_kernel<<<dim3(16 * 64), blk, 0, stream>>>(x, Wq, bq, Wk, bk, Wv, bv, q_rm, k_rm, v_rm);
  // 2) transposed copies for the column pass
  transpose_kernel<<<dim3(16 * 8 * 16), blk, 0, stream>>>(q_rm, q_cm);
  transpose_kernel<<<dim3(16 * 8 * 16), blk, 0, stream>>>(k_rm, k_cm);
  transpose_kernel<<<dim3(16 * 64 * 16), blk, 0, stream>>>(v_rm, v_cm);
  // 3) row pass (energy_W / out_W, no mask)
  attn_line_kernel<<<dim3(16 * 128), blk, 0, stream>>>(q_rm, k_rm, v_rm, oW, mW, sW, 0);
  // 4) column pass (energy_H / out_H, diagonal mask)
  attn_line_kernel<<<dim3(16 * 128), blk, 0, stream>>>(q_cm, k_cm, v_cm, oH, mH, sH, 1);
  // 5) merge softmax stats
  scales_kernel<<<dim3(1024), blk, 0, stream>>>(mW, sW, mH, sH, gamma, scW, scH);
  // 6) combine + residual
  combine_kernel<<<dim3(16 * 64 * 16), blk, 0, stream>>>(oW, oH, scW, scH, x, out);
}
